// TwistorLNN_37572373905541
// MI455X (gfx1250) — matrix operand order repaired and checked by reading, not compiled, not run on hardware
//
#include <hip/hip_runtime.h>
#include <stdint.h>

// ---------------- types for CDNA5 WMMA fragments ----------------
typedef __attribute__((ext_vector_type(16))) __bf16        v16bf;
typedef __attribute__((ext_vector_type(8)))  float         v8f;
typedef __attribute__((ext_vector_type(4)))  unsigned int  u32x4;

union FragAB {
  v16bf    v;
  u32x4    q[2];
  uint16_t s[16];
};

// ---------------- problem constants ----------------
constexpr int T_STEPS  = 1024;
constexpr int BATCH    = 256;
constexpr int IN_DIMC  = 128;
constexpr int HIDC     = 256;
constexpr int OUT_DIMC = 64;
constexpr int B_TILE   = 16;     // batch rows per workgroup
constexpr float DT_C   = 0.1f;
constexpr float EPS_C  = 1e-6f;

__device__ __forceinline__ uint16_t f32_to_bf16(float f) {
  union { float f; uint32_t u; } c; c.f = f;
  uint32_t u = c.u;
  uint32_t r = u + 0x7fffu + ((u >> 16) & 1u);   // round-to-nearest-even
  return (uint16_t)(r >> 16);
}

// Load one 16-element B fragment (16 contiguous K values of one W row),
// converting f32 -> bf16. Startup only; fragments then stay in VGPRs.
__device__ __forceinline__ v16bf load_b_frag_f32(const float* __restrict__ p) {
  FragAB f;
  #pragma unroll
  for (int j = 0; j < 16; ++j) f.s[j] = f32_to_bf16(p[j]);
  return f.v;
}

// ---------------- persistent scan kernel ----------------
// NOTE: the reference's imaginary state zi never influences y (y depends only
// on zr; tau depends only on zr; dzr never reads zi) -> the whole imag path is
// dead code w.r.t. the output and is eliminated.
//
// grid = BATCH/B_TILE = 16 workgroups, 256 threads (8 wave32 waves).
// W_tau/W_real B-fragments are operand-stationary in VGPRs (256 VGPRs/lane);
// U and W_out live in LDS; zr state lives in LDS (f32 + ping-pong bf16 stages).
__global__ __launch_bounds__(256, 1) void twistor_scan(
    const float* __restrict__ xg,
    const float* __restrict__ Wreal_g, const float* __restrict__ b_real,
    const float* __restrict__ Ug,      const float* __restrict__ b_U,
    const float* __restrict__ Wtau_g,  const float* __restrict__ b_tau,
    const float* __restrict__ Wout_g,  const float* __restrict__ b_out,
    float* __restrict__ yg)
{
  // ---- LDS: 148 KB total ----
  __shared__ __align__(16) float    zr_s[B_TILE * HIDC];        // 16 KB f32 state
  __shared__ __align__(16) uint16_t zrb [2][B_TILE * HIDC];     // 16 KB bf16(zr), ping-pong
  __shared__ __align__(16) uint16_t hr  [2][B_TILE * HIDC];     // 16 KB bf16(tanh(zr)), ping-pong
  __shared__ __align__(16) uint16_t xbuf[B_TILE * IN_DIMC];     //  4 KB bf16(x_t tile)
  __shared__ __align__(16) uint16_t Ub_s[HIDC * IN_DIMC];       // 64 KB bf16(U)
  __shared__ __align__(16) uint16_t Wo_s[OUT_DIMC * HIDC];      // 32 KB bf16(W_out)

  const int tid  = threadIdx.x;
  const int bg   = blockIdx.x;        // batch tile index
  const int wv   = tid >> 5;          // wave 0..7
  const int lane = tid & 31;
  const int nl   = lane & 15;         // A: M row / B,C,D: N col
  const int kh   = lane >> 4;         // K-half select per wave32 WMMA layout

  // ---- one-time staging ----
  for (int i = tid; i < B_TILE * HIDC; i += 256) {
    zr_s[i] = 0.f; zrb[0][i] = 0; hr[0][i] = 0;   // tanh(0) == 0
  }
  for (int i = tid; i < HIDC * IN_DIMC; i += 256)  Ub_s[i] = f32_to_bf16(Ug[i]);
  for (int i = tid; i < OUT_DIMC * HIDC; i += 256) Wo_s[i] = f32_to_bf16(Wout_g[i]);

  // ---- one-time: resident B fragments (wave wv owns N-tiles 2wv, 2wv+1) ----
  v16bf wT[2][8], wR[2][8];            // 32 frags * 8 VGPRs = 256 VGPRs/lane
  #pragma unroll
  for (int e = 0; e < 2; ++e) {
    const int n = (wv * 2 + e) * 16 + nl;
    #pragma unroll
    for (int kt = 0; kt < HIDC / 32; ++kt) {
      const int kb = kt * 32 + kh * 16;  // B layout: 16 contiguous K per lane
      wT[e][kt] = load_b_frag_f32(Wtau_g  + (size_t)n * HIDC + kb);
      wR[e][kt] = load_b_frag_f32(Wreal_g + (size_t)n * HIDC + kb);
    }
  }
  __syncthreads();

  for (int t = 0; t < T_STEPS; ++t) {
    const int cur = t & 1, nxt = cur ^ 1;

    // ---- Phase A: load x_t tile (only remaining prologue work) ----
    for (int i = tid; i < B_TILE * IN_DIMC; i += 256) {
      int m = i >> 7, k = i & (IN_DIMC - 1);
      const float* p = xg + ((size_t)t * BATCH + bg * B_TILE + m) * IN_DIMC + k;
      xbuf[i] = f32_to_bf16(*p);
      if (t + 1 < T_STEPS)
        __builtin_prefetch(p + (size_t)BATCH * IN_DIMC, 0, 0);  // global_prefetch_b8
    }
    __syncthreads();

    // ---- Phase B: tau & real GEMMs (resident B) + input projection + update ----
    #pragma unroll
    for (int e = 0; e < 2; ++e) {
      const int n = (wv * 2 + e) * 16 + nl;
      v8f acc_t = {}, acc_r = {}, acc_u = {};

      const uint16_t* arow_z = zrb[cur] + nl * HIDC;
      const uint16_t* arow_r = hr [cur] + nl * HIDC;

      #pragma unroll
      for (int kt = 0; kt < HIDC / 32; ++kt) {
        FragAB az, ah;
        const int ka = kt * 32 + kh * 16;  // A layout: 16 contiguous K per lane
        az.q[0] = *(const u32x4*)(arow_z + ka);
        az.q[1] = *(const u32x4*)(arow_z + ka + 8);
        ah.q[0] = *(const u32x4*)(arow_r + ka);
        ah.q[1] = *(const u32x4*)(arow_r + ka + 8);
        acc_t = __builtin_amdgcn_wmma_f32_16x16x32_bf16(false, az.v, false, wT[e][kt], (short)0, acc_t, false, false);
        acc_r = __builtin_amdgcn_wmma_f32_16x16x32_bf16(false, ah.v, false, wR[e][kt], (short)0, acc_r, false, false);
      }

      const uint16_t* arow_x = xbuf + nl * IN_DIMC;
      const uint16_t* brow_u = Ub_s + n * IN_DIMC;
      #pragma unroll
      for (int kt = 0; kt < IN_DIMC / 32; ++kt) {
        FragAB ax, bu;
        const int ka = kt * 32 + kh * 16;
        ax.q[0] = *(const u32x4*)(arow_x + ka);
        ax.q[1] = *(const u32x4*)(arow_x + ka + 8);
        const int kb = kt * 32 + kh * 16;
        bu.q[0] = *(const u32x4*)(brow_u + kb);
        bu.q[1] = *(const u32x4*)(brow_u + kb + 8);
        acc_u = __builtin_amdgcn_wmma_f32_16x16x32_bf16(false, ax.v, false, bu.v, (short)0, acc_u, false, false);
      }

      // Euler step on the C/D fragment; also emit next step's bf16 stages.
      // Each (m, n) element is owned by exactly one lane -> race-free writes;
      // reads come from the 'cur' ping-pong buffers, writes go to 'nxt'.
      const float btau = b_tau[n], brl = b_real[n], buu = b_U[n];
      #pragma unroll
      for (int v = 0; v < 8; ++v) {
        const int m   = v + kh * 8;
        const int idx = m * HIDC + n;
        const float tau = 1.f / (1.f + __expf(-(acc_t[v] + btau))) + EPS_C;
        float zrv = zr_s[idx];
        const float dzr = (-zrv + acc_r[v] + brl + acc_u[v] + buu) / tau;
        zrv += DT_C * dzr;
        zr_s[idx]    = zrv;
        zrb[nxt][idx] = f32_to_bf16(zrv);
        hr [nxt][idx] = f32_to_bf16(tanhf(zrv));
      }
    }
    __syncthreads();

    // ---- Phase C: y_t = zr_new @ W_out^T + b_out ; waves 0..3, one N-tile each ----
    if (wv < 4) {
      const int n = wv * 16 + nl;
      v8f acc = {};
      const uint16_t* arow = zrb[nxt] + nl * HIDC;
      const uint16_t* brow = Wo_s + n * HIDC;
      #pragma unroll
      for (int kt = 0; kt < HIDC / 32; ++kt) {
        FragAB a, b;
        const int ka = kt * 32 + kh * 16;
        a.q[0] = *(const u32x4*)(arow + ka);
        a.q[1] = *(const u32x4*)(arow + ka + 8);
        const int kb = kt * 32 + kh * 16;
        b.q[0] = *(const u32x4*)(brow + kb);
        b.q[1] = *(const u32x4*)(brow + kb + 8);
        acc = __builtin_amdgcn_wmma_f32_16x16x32_bf16(false, a.v, false, b.v, (short)0, acc, false, false);
      }
      const float bo = b_out[n];
      #pragma unroll
      for (int v = 0; v < 8; ++v) {
        const int m = v + kh * 8;
        yg[((size_t)t * BATCH + bg * B_TILE + m) * OUT_DIMC + n] = acc[v] + bo;
      }
    }
    // No barrier here: next Phase A writes only xbuf (untouched by Phase C);
    // the end-of-A barrier orders everything before zrb/hr are read/written.
  }
}

// ---------------- host-side launcher ----------------
extern "C" void kernel_launch(void* const* d_in, const int* in_sizes, int n_in,
                              void* d_out, int out_size, void* d_ws, size_t ws_size,
                              hipStream_t stream) {
  (void)in_sizes; (void)n_in; (void)out_size; (void)d_ws; (void)ws_size;
  const float* x      = (const float*)d_in[0];
  const float* W_real = (const float*)d_in[1];
  const float* b_real = (const float*)d_in[2];
  // d_in[3] = W_imag, d_in[4] = b_imag: provably dead w.r.t. the output y.
  const float* U      = (const float*)d_in[5];
  const float* b_U    = (const float*)d_in[6];
  const float* W_tau  = (const float*)d_in[7];
  const float* b_tau  = (const float*)d_in[8];
  const float* W_out  = (const float*)d_in[9];
  const float* b_out  = (const float*)d_in[10];

  twistor_scan<<<BATCH / B_TILE, 256, 0, stream>>>(
      x, W_real, b_real, U, b_U, W_tau, b_tau, W_out, b_out, (float*)d_out);
}
